// AtomicDielectricMACE_29729763623065
// MI455X (gfx1250) — compile-verified, matrix operand order checked by reading, not run on hardware
//
#include <hip/hip_runtime.h>
#include <hip/hip_bf16.h>
#include <math.h>

// Problem constants (from reference)
#define NN   16384
#define EE   262144
#define GG   64
#define CC   32
#define NEL  10
#define NBAS 8
#define NLAY 2

typedef __attribute__((ext_vector_type(16))) _Float16 v16h;
typedef __attribute__((ext_vector_type(8)))  float    v8f;

__device__ __forceinline__ int frag_k(int e, int hi) {
    // K index for f16 A/B fragments of v_wmma_f32_16x16x32_f16 (wave32):
    // element e of v16h, lane-half hi = lane>>4.
    return (e & 7) + ((e >> 3) << 4) + (hi << 3);
}

__device__ __forceinline__ float silu(float x) {
    // x * sigmoid(x) with hardware v_exp_f32 + v_rcp_f32 (no IEEE divide chain)
    return x * __builtin_amdgcn_rcpf(1.0f + __expf(-x));
}

#define WMMA16(A, B, Cacc) \
    __builtin_amdgcn_wmma_f32_16x16x32_f16(false, (A), false, (B), (short)0, (Cacc), false, false)

// ---------------------------------------------------------------------------
// Kernel 1: edge geometry -> sh (E x 16 f32), cutoff (E f32), and the radial
// basis stored directly in WMMA A-fragment order (16-edge tiles, K padded to
// 32 with zeros): ef_frag[(e>>4)*512 + lane*16 + elem], 32B-contig per lane.
// ---------------------------------------------------------------------------
__global__ __launch_bounds__(256) void k_geom(const int* __restrict__ ei,
                                              const float* __restrict__ pos,
                                              const float* __restrict__ shifts,
                                              float* __restrict__ sh,
                                              _Float16* __restrict__ ef_frag,
                                              float* __restrict__ cut) {
    int e = blockIdx.x * 256 + threadIdx.x;
    int s = ei[e], r = ei[EE + e];
    float vx = pos[r * 3 + 0] - pos[s * 3 + 0] + shifts[e * 3 + 0];
    float vy = pos[r * 3 + 1] - pos[s * 3 + 1] + shifts[e * 3 + 1];
    float vz = pos[r * 3 + 2] - pos[s * 3 + 2] + shifts[e * 3 + 2];
    float len = sqrtf(vx * vx + vy * vy + vz * vz + 1e-12f);
    float inv = 1.0f / len;
    float x = vx * inv, y = vy * inv, z = vz * inv;
    float x2 = x * x, y2 = y * y, z2 = z * z;

    const float s3 = 1.7320508075688772f, s15 = 3.872983346207417f;
    const float c6 = 1.118033988749895f;       // sqrt(5)/2
    const float c9 = 2.091650066335189f;       // sqrt(35/8)
    const float c10 = 10.246950765959598f;     // sqrt(105)
    const float c11 = 1.6201851746019651f;     // sqrt(21/8)
    const float c12 = 1.3228756555322954f;     // sqrt(7)/2
    const float c14 = 5.123475382979799f;      // sqrt(105)/2

    float* sp = sh + (size_t)e * 16;
    sp[0]  = 1.0f;
    sp[1]  = s3 * x;  sp[2] = s3 * y;  sp[3] = s3 * z;
    sp[4]  = s15 * x * y;
    sp[5]  = s15 * y * z;
    sp[6]  = c6 * (3.0f * z2 - 1.0f);
    sp[7]  = s15 * x * z;
    sp[8]  = 0.5f * s15 * (x2 - y2);
    sp[9]  = c9 * y * (3.0f * x2 - y2);
    sp[10] = c10 * x * y * z;
    sp[11] = c11 * y * (5.0f * z2 - 1.0f);
    sp[12] = c12 * z * (5.0f * z2 - 3.0f);
    sp[13] = c11 * x * (5.0f * z2 - 1.0f);
    sp[14] = c14 * z * (x2 - y2);
    sp[15] = c9 * x * (x2 - 3.0f * y2);

    float t = len * 0.2f;  // R_MAX = 5
    float t2 = t * t, t4 = t2 * t2, t5 = t4 * t;
    float env = 1.0f - 21.0f * t5 + 35.0f * t5 * t - 15.0f * t5 * t2;
    env = (t < 1.0f) ? env : 0.0f;
    cut[e] = env;

    const float pref = 0.6324555320336759f;  // sqrt(2/5)
    // lo-lane chunk: elem j<8 -> K=j (value); j>=8 -> K=16+j-8 (zero pad)
    v16h lo;
#pragma unroll
    for (int j = 0; j < 16; ++j) lo[j] = (_Float16)0.0f;
#pragma unroll
    for (int j = 0; j < NBAS; ++j) {
        float nv = (float)(j + 1);
        float rb = pref * __sinf(nv * 3.14159265358979f * len * 0.2f) * inv;
        lo[j] = (_Float16)(rb * env);
    }
    v16h zz;
#pragma unroll
    for (int j = 0; j < 16; ++j) zz[j] = (_Float16)0.0f;

    _Float16* base = ef_frag + (size_t)(e >> 4) * 512;
    int lane = e & 15;
    *(v16h*)(base + lane * 16) = lo;          // lanes 0..15: K in {0..7,16..23}
    *(v16h*)(base + (16 + lane) * 16) = zz;   // lanes 16..31: K>=8 -> all zero
}

// ---------------------------------------------------------------------------
// Kernel 2: fused 4-layer edge MLP via WMMA f16 (f32 accum).
// 8 waves/block, 16 edges/wave, 128 edges/block, 36 WMMA/wave.
// Weights pre-swizzled into fragment order in LDS: B-frag = one contiguous
// 32B chunk per lane -> 2x ds_load_b128 instead of 16x ds_load_u16.
// Activations round-trip LDS in A-fragment order.
// ---------------------------------------------------------------------------
__global__ __launch_bounds__(256) void k_edge_mlp(int layer,
                                                  const _Float16* __restrict__ ef_frag,
                                                  const float* __restrict__ cut,
                                                  const float* __restrict__ Wr1,
                                                  const float* __restrict__ br1,
                                                  const float* __restrict__ Wr2,
                                                  const float* __restrict__ Wr3,
                                                  const float* __restrict__ Wro,
                                                  _Float16* __restrict__ wout) {
    // fragment-order layout: [(t*KB + kk)*32 + lane]*16 + elem
    __shared__ alignas(32) _Float16 lW1f[4 * 1 * 32 * 16];   //  4 KB
    __shared__ float lB1[64];
    __shared__ alignas(32) _Float16 lW2f[4 * 2 * 32 * 16];   //  8 KB
    __shared__ alignas(32) _Float16 lW3f[4 * 2 * 32 * 16];   //  8 KB
    __shared__ alignas(32) _Float16 lWof[8 * 2 * 32 * 16];   // 16 KB
    __shared__ alignas(32) _Float16 lH[8][2 * 32 * 16];      // 16 KB, A-frag order

    int tid = threadIdx.x;
    // ---- stage weights, swizzled to fragment order ----
    for (int idx = tid; idx < 2048; idx += 256) {            // W1: 32x64, KB=1
        int el = idx & 15, l = (idx >> 4) & 31, t = idx >> 9;
        int k = frag_k(el, l >> 4);
        int col = t * 16 + (l & 15);
        lW1f[idx] = (_Float16)(k < NBAS ? Wr1[(layer * NBAS + k) * 64 + col] : 0.0f);
    }
    if (tid < 64) lB1[tid] = br1[layer * 64 + tid];
    for (int idx = tid; idx < 4096; idx += 256) {            // W2/W3: 64x64, KB=2
        int el = idx & 15, l = (idx >> 4) & 31, blk = idx >> 9;
        int kk = blk & 1, t = blk >> 1;
        int k = kk * 32 + frag_k(el, l >> 4);
        int col = t * 16 + (l & 15);
        lW2f[idx] = (_Float16)Wr2[layer * 4096 + k * 64 + col];
        lW3f[idx] = (_Float16)Wr3[layer * 4096 + k * 64 + col];
    }
    for (int idx = tid; idx < 8192; idx += 256) {            // Wout: 64x128, KB=2
        int el = idx & 15, l = (idx >> 4) & 31, blk = idx >> 9;
        int kk = blk & 1, t = blk >> 1;
        int k = kk * 32 + frag_k(el, l >> 4);
        int col = t * 16 + (l & 15);
        lWof[idx] = (_Float16)Wro[layer * 8192 + k * 128 + col];
    }
    __syncthreads();

    int wave = tid >> 5, lane = tid & 31;
    int hi = lane >> 4, ncol = lane & 15;
    size_t e0 = (size_t)blockIdx.x * 128 + (size_t)wave * 16;
    _Float16* H = lH[wave];

    // address pieces for writing activations into A-fragment order:
    // value at (row M, col K): frag f=K>>5, lane=(M&15)+16*((K>>3)&1),
    // elem=(K&7)+8*((K>>4)&1).  Per tile t: K = t*16 + ncol.
    int w_khi = (ncol >> 3) & 1;          // (K>>3)&1, independent of t
    int w_k7 = ncol & 7;                  // K&7

    // ----- layer 1: (16x8 pad->32) @ (32x64), bias, silu -----
    v16h a1 = *(const v16h*)(ef_frag + (e0 >> 4) * 512 + lane * 16);
#pragma unroll
    for (int t = 0; t < 4; ++t) {
        v16h b = *(const v16h*)(lW1f + (t * 32 + lane) * 16);
        v8f acc;
        float bias = lB1[t * 16 + ncol];
#pragma unroll
        for (int j = 0; j < 8; ++j) acc[j] = bias;
        acc = WMMA16(a1, b, acc);
        int f = t >> 1;
        int eidx = w_k7 + 8 * (t & 1);
#pragma unroll
        for (int j = 0; j < 8; ++j) {
            int M = j + 8 * hi;
            H[(f * 32 + M + 16 * w_khi) * 16 + eidx] = (_Float16)silu(acc[j]);
        }
    }

    // ----- layers 2 and 3: (16x64) @ (64x64), silu, in-place via LDS -----
#pragma unroll
    for (int L = 0; L < 2; ++L) {
        const _Float16* W = (L == 0) ? lW2f : lW3f;
        v16h alo = *(const v16h*)(H + lane * 16);
        v16h ahi = *(const v16h*)(H + (32 + lane) * 16);
#pragma unroll
        for (int t = 0; t < 4; ++t) {
            v16h blo = *(const v16h*)(W + ((t * 2 + 0) * 32 + lane) * 16);
            v16h bhi = *(const v16h*)(W + ((t * 2 + 1) * 32 + lane) * 16);
            v8f acc;
#pragma unroll
            for (int j = 0; j < 8; ++j) acc[j] = 0.0f;
            acc = WMMA16(alo, blo, acc);
            acc = WMMA16(ahi, bhi, acc);
            int f = t >> 1;
            int eidx = w_k7 + 8 * (t & 1);
#pragma unroll
            for (int j = 0; j < 8; ++j) {
                int M = j + 8 * hi;
                H[(f * 32 + M + 16 * w_khi) * 16 + eidx] = (_Float16)silu(acc[j]);
            }
        }
    }

    // ----- output layer: (16x64) @ (64x128), scale by cutoff, store f16 -----
    float cutv[8];
#pragma unroll
    for (int j = 0; j < 8; ++j) cutv[j] = cut[e0 + j + 8 * hi];
    v16h alo = *(const v16h*)(H + lane * 16);
    v16h ahi = *(const v16h*)(H + (32 + lane) * 16);
#pragma unroll
    for (int t = 0; t < 8; ++t) {
        v16h blo = *(const v16h*)(lWof + ((t * 2 + 0) * 32 + lane) * 16);
        v16h bhi = *(const v16h*)(lWof + ((t * 2 + 1) * 32 + lane) * 16);
        v8f acc;
#pragma unroll
        for (int j = 0; j < 8; ++j) acc[j] = 0.0f;
        acc = WMMA16(alo, blo, acc);
        acc = WMMA16(ahi, bhi, acc);
#pragma unroll
        for (int j = 0; j < 8; ++j)
            wout[(e0 + j + 8 * hi) * 128 + t * 16 + ncol] = (_Float16)(acc[j] * cutv[j]);
    }
}

// ---------------------------------------------------------------------------
// Kernel 3: message formation + scatter (hardware f32 atomics into agg)
// thread = (edge, channel); 16 atomic adds each.
// ---------------------------------------------------------------------------
__global__ __launch_bounds__(256) void k_scatter(const int* __restrict__ ei,
                                                 const float* __restrict__ feats,
                                                 const _Float16* __restrict__ w,
                                                 const float* __restrict__ sh,
                                                 float* __restrict__ agg) {
    size_t idx = (size_t)blockIdx.x * 256 + threadIdx.x;  // EE*32 threads
    int e = (int)(idx >> 5);
    int c = (int)(idx & 31);
    int snd = ei[e], rcv = ei[EE + e];
    float s = feats[(size_t)snd * 512 + c * 16 + 0];
    const _Float16* wp = w + (size_t)e * 128 + c * 4;
    float wv[4] = {(float)wp[0], (float)wp[1], (float)wp[2], (float)wp[3]};
    const float* shp = sh + (size_t)e * 16;
    float* ap = agg + (size_t)rcv * 512 + c * 16;
    const int LIDX[16] = {0, 1, 1, 1, 2, 2, 2, 2, 2, 3, 3, 3, 3, 3, 3, 3};
#pragma unroll
    for (int l = 0; l < 16; ++l) {
        float v = s * wv[LIDX[l]] * shp[l] * 0.0625f;  // / AVG_NEIGH
        __hip_atomic_fetch_add(ap + l, v, __ATOMIC_RELAXED, __HIP_MEMORY_SCOPE_AGENT);
    }
}

// ---------------------------------------------------------------------------
// Kernel 4: per-node mixes + gating + residual + readout accumulation
// one block (256 threads) per node.
// ---------------------------------------------------------------------------
__global__ __launch_bounds__(256) void k_node(int layer,
                                              const float* __restrict__ agg,
                                              const float* __restrict__ fin,
                                              const float* __restrict__ attrs,
                                              const float* __restrict__ W_mix,
                                              const float* __restrict__ W_sc,
                                              const float* __restrict__ Wp2,
                                              const float* __restrict__ Wp3,
                                              const float* __restrict__ W_plin,
                                              const float* __restrict__ W_read,
                                              float* __restrict__ fout,
                                              float* __restrict__ chg,
                                              float* __restrict__ dip,
                                              float* __restrict__ polar) {
    __shared__ float sAgg[512], sIn[512], sM[512], sSc[512], sB[512], sNew[512];
    int n = blockIdx.x, t = threadIdx.x;
    const int LIDX[16] = {0, 1, 1, 1, 2, 2, 2, 2, 2, 3, 3, 3, 3, 3, 3, 3};

    sAgg[t] = agg[(size_t)n * 512 + t];
    sAgg[t + 256] = agg[(size_t)n * 512 + t + 256];
    sIn[t] = fin[(size_t)n * 512 + t];
    sIn[t + 256] = fin[(size_t)n * 512 + t + 256];
    __syncthreads();

#pragma unroll
    for (int r = 0; r < 2; ++r) {
        int o = t + 256 * r;
        int d = o >> 4, m = o & 15, l = LIDX[m];
        const float* wm = W_mix + (size_t)((layer * 4 + l) * 32) * 32 + d;
        const float* ws = W_sc + (size_t)((layer * 4 + l) * 32) * 32 + d;
        float am = 0.0f, as = 0.0f;
        for (int c = 0; c < 32; ++c) {
            am += sAgg[c * 16 + m] * wm[c * 32];
            as += sIn[c * 16 + m] * ws[c * 32];
        }
        sM[o] = am;
        sSc[o] = as;
    }
    __syncthreads();

#pragma unroll
    for (int r = 0; r < 2; ++r) {
        int o = t + 256 * r;
        int d = o >> 4;
        float g2 = 0.0f, g3 = 0.0f;
        for (int k = 0; k < NEL; ++k) {
            float a = attrs[(size_t)n * NEL + k];
            g2 += a * Wp2[(layer * NEL + k) * 32 + d];
            g3 += a * Wp3[(layer * NEL + k) * 32 + d];
        }
        float sn = sM[d * 16];
        sB[o] = sM[o] * (1.0f + g2 * sn + g3 * sn * sn);
    }
    __syncthreads();

#pragma unroll
    for (int r = 0; r < 2; ++r) {
        int o = t + 256 * r;
        int d = o >> 4, m = o & 15, l = LIDX[m];
        const float* wp = W_plin + (size_t)((layer * 4 + l) * 32) * 32 + d;
        float acc = 0.0f;
        for (int c = 0; c < 32; ++c) acc += sB[c * 16 + m] * wp[c * 32];
        sNew[o] = acc + sSc[o];
    }
    __syncthreads();

    fout[(size_t)n * 512 + t] = sNew[t];
    fout[(size_t)n * 512 + t + 256] = sNew[t + 256];

    if (t < 10) {
        if (t == 0) {
            float acc = 0.0f;
            for (int d = 0; d < 32; ++d) {
                float g = sNew[d * 16];
                if (layer == NLAY - 1) g = silu(g);
                acc += g * W_read[(layer * 4 + 0) * 32 + d];
            }
            chg[n] += acc;
        } else if (t == 1) {
            float acc = 0.0f;
            for (int d = 0; d < 32; ++d) {
                float g = sNew[d * 16];
                if (layer == NLAY - 1) g = silu(g);
                acc += g * W_read[(layer * 4 + 1) * 32 + d];
            }
            polar[(size_t)n * 6 + 0] += acc;
        } else if (t < 5) {
            int m = t - 2;
            float acc = 0.0f;
            for (int d = 0; d < 32; ++d)
                acc += sNew[d * 16 + 1 + m] * W_read[(layer * 4 + 2) * 32 + d];
            dip[(size_t)n * 3 + m] += acc;
        } else {
            int m = t - 5;
            float acc = 0.0f;
            for (int d = 0; d < 32; ++d)
                acc += sNew[d * 16 + 4 + m] * W_read[(layer * 4 + 3) * 32 + d];
            polar[(size_t)n * 6 + 1 + m] += acc;
        }
    }
}

// ---------------------------------------------------------------------------
// Init kernels
// ---------------------------------------------------------------------------
__global__ __launch_bounds__(256) void k_init_feats(const float* __restrict__ attrs,
                                                    const float* __restrict__ W_embed,
                                                    float* __restrict__ feats) {
    size_t o = (size_t)blockIdx.x * 256 + threadIdx.x;  // NN*512
    int n = (int)(o >> 9);
    int rem = (int)(o & 511);
    int c = rem >> 4, m = rem & 15;
    float v = 0.0f;
    if (m == 0) {
        for (int k = 0; k < NEL; ++k) v += attrs[(size_t)n * NEL + k] * W_embed[k * 32 + c];
    }
    feats[o] = v;
}

__global__ __launch_bounds__(256) void k_init_accum(float* chg, float* dip, float* polar) {
    int n = blockIdx.x * 256 + threadIdx.x;
    chg[n] = 0.0f;
    for (int j = 0; j < 3; ++j) dip[(size_t)n * 3 + j] = 0.0f;
    for (int j = 0; j < 6; ++j) polar[(size_t)n * 6 + j] = 0.0f;
}

// ---------------------------------------------------------------------------
// Kernel 5: per-graph reductions, charge neutralization, COB transform
// one block per graph; exactly 256 nodes/graph.
// ---------------------------------------------------------------------------
__device__ float blk_reduce(float v, float* red, int t) {
    red[t] = v;
    __syncthreads();
    for (int s = 128; s > 0; s >>= 1) {
        if (t < s) red[t] += red[t + s];
        __syncthreads();
    }
    float r = red[0];
    __syncthreads();
    return r;
}

__global__ __launch_bounds__(256) void k_finalize(const float* __restrict__ chg,
                                                  const float* __restrict__ dip,
                                                  const float* __restrict__ polar,
                                                  const float* __restrict__ pos,
                                                  const float* __restrict__ total_charge,
                                                  float* __restrict__ out) {
    __shared__ float red[256];
    int g = blockIdx.x, t = threadIdx.x;
    int n = g * 256 + t;
    float q = chg[n];
    float qsum = blk_reduce(q, red, t);
    float excess = qsum * (1.0f / 256.0f) - total_charge[g] * (1.0f / 256.0f);
    q -= excess;
    float px = pos[(size_t)n * 3 + 0], py = pos[(size_t)n * 3 + 1], pz = pos[(size_t)n * 3 + 2];
    float d0 = blk_reduce(dip[(size_t)n * 3 + 0] + q * px, red, t);
    float d1 = blk_reduce(dip[(size_t)n * 3 + 1] + q * py, red, t);
    float d2 = blk_reduce(dip[(size_t)n * 3 + 2] + q * pz, red, t);
    float p[6];
    for (int k = 0; k < 6; ++k) p[k] = blk_reduce(polar[(size_t)n * 6 + k], red, t);
    if (t == 0) {
        out[g * 12 + 0] = d0;
        out[g * 12 + 1] = d1;
        out[g * 12 + 2] = d2;
        const float is3 = 0.5773502691896258f, is2 = 0.7071067811865475f, is6 = 0.4082482904638631f;
        float M[9];
        M[0] = p[0] * is3 - p[3] * is6 + p[5] * is2;
        M[4] = p[0] * is3 - p[3] * is6 - p[5] * is2;
        M[8] = p[0] * is3 + 2.0f * p[3] * is6;
        M[1] = M[3] = p[1] * is2;
        M[5] = M[7] = p[2] * is2;
        M[2] = M[6] = p[4] * is2;
        for (int j = 0; j < 9; ++j) out[g * 12 + 3 + j] = M[j];
    }
}

// ---------------------------------------------------------------------------
// Host-side launch
// ---------------------------------------------------------------------------
extern "C" void kernel_launch(void* const* d_in, const int* in_sizes, int n_in,
                              void* d_out, int out_size, void* d_ws, size_t ws_size,
                              hipStream_t stream) {
    const float* node_attrs = (const float*)d_in[0];
    const float* positions  = (const float*)d_in[1];
    const float* shifts     = (const float*)d_in[2];
    const float* total_chg  = (const float*)d_in[3];
    const float* W_embed    = (const float*)d_in[4];
    const float* Wr1        = (const float*)d_in[5];
    const float* br1        = (const float*)d_in[6];
    const float* Wr2        = (const float*)d_in[7];
    const float* Wr3        = (const float*)d_in[8];
    const float* Wr_out     = (const float*)d_in[9];
    const float* W_mix      = (const float*)d_in[10];
    const float* W_sc       = (const float*)d_in[11];
    const float* Wp2        = (const float*)d_in[12];
    const float* Wp3        = (const float*)d_in[13];
    const float* W_plin     = (const float*)d_in[14];
    const float* W_read     = (const float*)d_in[15];
    const int*   edge_index = (const int*)d_in[16];
    float* out = (float*)d_out;

    // workspace carve-up (256B aligned)
    char* base = (char*)d_ws;
    size_t off = 0;
    auto carve = [&](size_t bytes) {
        char* p = base + off;
        off = (off + bytes + 255) & ~(size_t)255;
        return p;
    };
    float*    sh      = (float*)carve((size_t)EE * 16 * 4);
    _Float16* ef_frag = (_Float16*)carve((size_t)EE * 32 * 2);   // A-frag order
    float*    cut     = (float*)carve((size_t)EE * 4);
    _Float16* wbuf    = (_Float16*)carve((size_t)EE * 128 * 2);
    float*    agg     = (float*)carve((size_t)NN * 512 * 4);
    float*    feats0  = (float*)carve((size_t)NN * 512 * 4);
    float*    feats1  = (float*)carve((size_t)NN * 512 * 4);
    float*    chg     = (float*)carve((size_t)NN * 4);
    float*    dip     = (float*)carve((size_t)NN * 3 * 4);
    float*    polar   = (float*)carve((size_t)NN * 6 * 4);

    k_geom<<<EE / 256, 256, 0, stream>>>(edge_index, positions, shifts, sh, ef_frag, cut);
    k_init_feats<<<(NN * 512) / 256, 256, 0, stream>>>(node_attrs, W_embed, feats0);
    k_init_accum<<<NN / 256, 256, 0, stream>>>(chg, dip, polar);

    for (int layer = 0; layer < NLAY; ++layer) {
        const float* fin = (layer == 0) ? feats0 : feats1;
        float* fout = (layer == 0) ? feats1 : feats0;
        k_edge_mlp<<<EE / 128, 256, 0, stream>>>(layer, ef_frag, cut, Wr1, br1, Wr2, Wr3,
                                                 Wr_out, wbuf);
        hipMemsetAsync(agg, 0, (size_t)NN * 512 * 4, stream);
        k_scatter<<<(EE * 32) / 256, 256, 0, stream>>>(edge_index, fin, wbuf, sh, agg);
        k_node<<<NN, 256, 0, stream>>>(layer, agg, fin, node_attrs, W_mix, W_sc, Wp2, Wp3,
                                       W_plin, W_read, fout, chg, dip, polar);
    }

    k_finalize<<<GG, 256, 0, stream>>>(chg, dip, polar, positions, total_chg, out);
    (void)in_sizes; (void)n_in; (void)out_size; (void)ws_size;
}
